// BiDirectionalAttention_58085137711395
// MI455X (gfx1250) — matrix operand order repaired and checked by reading, not compiled, not run on hardware
//
#include <hip/hip_runtime.h>

// ---------------------------------------------------------------------------
// BiDirectionalAttention for MI455X (gfx1250, wave32, WMMA bf16 -> f32)
// C_LEN=8192, Q_LEN=1024, D=1024
// GEMMs use 64x64 wave tiles (16 WMMAs per 16 b128 loads -> 1 load/WMMA).
// ---------------------------------------------------------------------------

#define C_LEN 8192
#define Q_LEN 1024
#define D_DIM 1024

typedef __attribute__((ext_vector_type(16))) __bf16 v16bf;
typedef __attribute__((ext_vector_type(8)))  __bf16 v8bf;
typedef __attribute__((ext_vector_type(8)))  float  v8f;

// float -> bf16 (round to nearest even)
static __device__ __forceinline__ unsigned short f2bf(float f) {
    unsigned u = __float_as_uint(f);
    u += 0x7FFFu + ((u >> 16) & 1u);
    return (unsigned short)(u >> 16);
}

// Load a 16-half WMMA fragment as two b128 loads (lo 8 halfs at p, hi 8 at p+hiOff)
static __device__ __forceinline__ v16bf load_frag16(const unsigned short* p, int hiOff) {
    v8bf lo = *reinterpret_cast<const v8bf*>(p);
    v8bf hi = *reinterpret_cast<const v8bf*>(p + hiOff);
    return __builtin_shufflevector(lo, hi, 0,1,2,3,4,5,6,7,8,9,10,11,12,13,14,15);
}

static __device__ __forceinline__ v8f wmma_bf16(v16bf a, v16bf b, v8f c) {
    // D = A(16x32) * B(32x16) + C, f32 accumulate
    return __builtin_amdgcn_wmma_f32_16x16x32_bf16(false, a, false, b, (short)0, c, false, false);
}

// ---------------------------------------------------------------------------
// Prep H: Hs = bf16(H * w_qc), c_term = H @ w_c + b_c, rowb = rowmax(H)
// One wave per row; 1024 blocks x 256 threads.
// ---------------------------------------------------------------------------
__global__ __launch_bounds__(256) void prep_h_kernel(
    const float* __restrict__ H, const float* __restrict__ w_qc,
    const float* __restrict__ w_c, const float* __restrict__ b_c,
    unsigned short* __restrict__ Hs, float* __restrict__ c_term,
    float* __restrict__ rowb)
{
    const int wave = threadIdx.x >> 5, lane = threadIdx.x & 31;
    const int row  = blockIdx.x * 8 + wave;
    const float* h = H + (size_t)row * D_DIM;
    unsigned short* hs = Hs + (size_t)row * D_DIM;
    float dot = 0.f, mx = -1e30f;
    for (int d = lane * 4; d < D_DIM; d += 128) {
        float4 hv = *(const float4*)(h + d);
        float4 wq = *(const float4*)(w_qc + d);
        float4 wc = *(const float4*)(w_c + d);
        dot += hv.x*wc.x + hv.y*wc.y + hv.z*wc.z + hv.w*wc.w;
        mx = fmaxf(mx, fmaxf(fmaxf(hv.x, hv.y), fmaxf(hv.z, hv.w)));
        ushort4 o;
        o.x = f2bf(hv.x * wq.x); o.y = f2bf(hv.y * wq.y);
        o.z = f2bf(hv.z * wq.z); o.w = f2bf(hv.w * wq.w);
        *(ushort4*)(hs + d) = o;
    }
    for (int off = 16; off; off >>= 1) {
        dot += __shfl_xor(dot, off, 32);
        mx   = fmaxf(mx, __shfl_xor(mx, off, 32));
    }
    if (lane == 0) { c_term[row] = dot + b_c[0]; rowb[row] = mx; }
}

// ---------------------------------------------------------------------------
// Prep U: Ub = bf16(U) row-major, Ut = bf16(U)^T, q_term = U @ w_q + b_q
// One wave per row; 128 blocks x 256 threads.
// ---------------------------------------------------------------------------
__global__ __launch_bounds__(256) void prep_u_kernel(
    const float* __restrict__ U, const float* __restrict__ w_q,
    const float* __restrict__ b_q,
    unsigned short* __restrict__ Ub, unsigned short* __restrict__ Ut,
    float* __restrict__ q_term)
{
    const int wave = threadIdx.x >> 5, lane = threadIdx.x & 31;
    const int q = blockIdx.x * 8 + wave;
    const float* u = U + (size_t)q * D_DIM;
    float dot = 0.f;
    for (int d = lane * 4; d < D_DIM; d += 128) {
        float4 uv = *(const float4*)(u + d);
        float4 wv = *(const float4*)(w_q + d);
        dot += uv.x*wv.x + uv.y*wv.y + uv.z*wv.z + uv.w*wv.w;
        ushort4 o;
        o.x = f2bf(uv.x); o.y = f2bf(uv.y); o.z = f2bf(uv.z); o.w = f2bf(uv.w);
        *(ushort4*)(Ub + (size_t)q * D_DIM + d) = o;
        Ut[(size_t)(d + 0) * Q_LEN + q] = o.x;
        Ut[(size_t)(d + 1) * Q_LEN + q] = o.y;
        Ut[(size_t)(d + 2) * Q_LEN + q] = o.z;
        Ut[(size_t)(d + 3) * Q_LEN + q] = o.w;
    }
    for (int off = 16; off; off >>= 1) dot += __shfl_xor(dot, off, 32);
    if (lane == 0) q_term[q] = dot + b_q[0];
}

// ---------------------------------------------------------------------------
// GEMM1: S[c,q] = Hs @ Ub^T + c_term[c] + q_term[q] + b_qc
// Block = 256 thr (8 waves as 4x2), block tile 256x128, wave tile 64x64.
// Grid (32, 8).
// ---------------------------------------------------------------------------
__global__ __launch_bounds__(256) void gemm_s_kernel(
    const unsigned short* __restrict__ A,  // Hs [C_LEN, D]
    const unsigned short* __restrict__ B,  // Ub [Q_LEN, D] (row n = B column n)
    const float* __restrict__ c_term, const float* __restrict__ q_term,
    const float* __restrict__ b_qc, float* __restrict__ S)
{
    const int wave = threadIdx.x >> 5, lane = threadIdx.x & 31;
    const int llo = lane & 15, lhi = lane >> 4;
    const int m0 = blockIdx.x * 256 + (wave >> 1) * 64;
    const int n0 = blockIdx.y * 128 + (wave & 1) * 64;

    // Per-lane fragment base pointers matching ISA VGPR layouts.
    const unsigned short* aBase[4];
    const unsigned short* bBase[4];
    #pragma unroll
    for (int i = 0; i < 4; ++i) {
        aBase[i] = A + (size_t)(m0 + i * 16 + llo) * D_DIM + lhi * 8;
        bBase[i] = B + (size_t)(n0 + i * 16 + llo) * D_DIM + lhi * 8;
    }

    v8f acc[4][4] = {};
    for (int k = 0; k < D_DIM; k += 32) {
        __builtin_prefetch(aBase[0] + k + 256, 0, 3);
        __builtin_prefetch(bBase[0] + k + 256, 0, 3);
        v16bf fa[4], fb[4];
        #pragma unroll
        for (int i = 0; i < 4; ++i) fa[i] = load_frag16(aBase[i] + k, 16);
        #pragma unroll
        for (int j = 0; j < 4; ++j) fb[j] = load_frag16(bBase[j] + k, 16);
        #pragma unroll
        for (int i = 0; i < 4; ++i)
            #pragma unroll
            for (int j = 0; j < 4; ++j)
                acc[i][j] = wmma_bf16(fa[i], fb[j], acc[i][j]);
    }
    const float bias = b_qc[0];
    #pragma unroll
    for (int i = 0; i < 4; ++i) {
        #pragma unroll
        for (int j = 0; j < 4; ++j) {
            const int n = n0 + j * 16 + llo;
            const float qt = q_term[n] + bias;
            #pragma unroll
            for (int r = 0; r < 8; ++r) {
                const int m = m0 + i * 16 + lhi * 8 + r;
                S[(size_t)m * Q_LEN + n] = acc[i][j][r] + c_term[m] + qt;
            }
        }
    }
}

// ---------------------------------------------------------------------------
// Column softmax stats over the 8192 context rows (online max/sum merge).
// Block handles 16 columns; grid 64 x 256 threads.
// ---------------------------------------------------------------------------
__global__ __launch_bounds__(256) void colstats_kernel(
    const float* __restrict__ S, float* __restrict__ mq, float* __restrict__ rzq)
{
    const int tx = threadIdx.x & 15;   // column in group
    const int ty = threadIdx.x >> 4;   // row stripe 0..15
    const int q  = blockIdx.x * 16 + tx;
    float m = -1e30f, z = 0.f;
    for (int r = ty; r < C_LEN; r += 16) {
        float x = S[(size_t)r * Q_LEN + q];
        float mN = fmaxf(m, x);
        z = z * __expf(m - mN) + __expf(x - mN);
        m = mN;
    }
    __shared__ float sm[256], sz[256];
    sm[threadIdx.x] = m; sz[threadIdx.x] = z;
    __syncthreads();
    for (int s = 128; s >= 16; s >>= 1) {
        if (threadIdx.x < s) {
            float m2 = sm[threadIdx.x + s], z2 = sz[threadIdx.x + s];
            float mN = fmaxf(sm[threadIdx.x], m2);
            sz[threadIdx.x] = sz[threadIdx.x] * __expf(sm[threadIdx.x] - mN)
                            + z2 * __expf(m2 - mN);
            sm[threadIdx.x] = mN;
        }
        __syncthreads();
    }
    if (ty == 0) { mq[q] = sm[tx]; rzq[q] = 1.0f / sz[tx]; }
}

// ---------------------------------------------------------------------------
// P[c,q] = bf16( exp(S - m_q) * (1/z_q) ); 8192 blocks x 256 thr x 4 elems.
// ---------------------------------------------------------------------------
__global__ __launch_bounds__(256) void make_p_kernel(
    const float* __restrict__ S, const float* __restrict__ mq,
    const float* __restrict__ rzq, unsigned short* __restrict__ P)
{
    const size_t i = ((size_t)blockIdx.x * blockDim.x + threadIdx.x) * 4;
    const int q = (int)(i & (Q_LEN - 1));
    float4 s  = *(const float4*)(S + i);
    float4 m  = *(const float4*)(mq + q);
    float4 rz = *(const float4*)(rzq + q);
    ushort4 o;
    o.x = f2bf(__expf(s.x - m.x) * rz.x);
    o.y = f2bf(__expf(s.y - m.y) * rz.y);
    o.z = f2bf(__expf(s.z - m.z) * rz.z);
    o.w = f2bf(__expf(s.w - m.w) * rz.w);
    *(ushort4*)(P + i) = o;
}

// ---------------------------------------------------------------------------
// GEMM2: U_toggler[c,d] = P @ U  (B = Ut row-major so B[k,n]=Ut[n*Q+k]).
// Same 64x64 wave tiling; writes directly to d_out. Grid (32, 8).
// ---------------------------------------------------------------------------
__global__ __launch_bounds__(256) void gemm_out_kernel(
    const unsigned short* __restrict__ A,  // P  [C_LEN, Q_LEN]
    const unsigned short* __restrict__ B,  // Ut [D,    Q_LEN]
    float* __restrict__ out)
{
    const int wave = threadIdx.x >> 5, lane = threadIdx.x & 31;
    const int llo = lane & 15, lhi = lane >> 4;
    const int m0 = blockIdx.x * 256 + (wave >> 1) * 64;
    const int n0 = blockIdx.y * 128 + (wave & 1) * 64;

    const unsigned short* aBase[4];
    const unsigned short* bBase[4];
    #pragma unroll
    for (int i = 0; i < 4; ++i) {
        aBase[i] = A + (size_t)(m0 + i * 16 + llo) * Q_LEN + lhi * 8;
        bBase[i] = B + (size_t)(n0 + i * 16 + llo) * Q_LEN + lhi * 8;
    }

    v8f acc[4][4] = {};
    for (int k = 0; k < Q_LEN; k += 32) {
        __builtin_prefetch(aBase[0] + k + 256, 0, 3);
        __builtin_prefetch(bBase[0] + k + 256, 0, 3);
        v16bf fa[4], fb[4];
        #pragma unroll
        for (int i = 0; i < 4; ++i) fa[i] = load_frag16(aBase[i] + k, 16);
        #pragma unroll
        for (int j = 0; j < 4; ++j) fb[j] = load_frag16(bBase[j] + k, 16);
        #pragma unroll
        for (int i = 0; i < 4; ++i)
            #pragma unroll
            for (int j = 0; j < 4; ++j)
                acc[i][j] = wmma_bf16(fa[i], fb[j], acc[i][j]);
    }
    #pragma unroll
    for (int i = 0; i < 4; ++i)
        #pragma unroll
        for (int j = 0; j < 4; ++j) {
            const int n = n0 + j * 16 + llo;
            #pragma unroll
            for (int r = 0; r < 8; ++r) {
                const int m = m0 + i * 16 + lhi * 8 + r;
                out[(size_t)m * D_DIM + n] = acc[i][j][r];
            }
        }
}

// ---------------------------------------------------------------------------
// c2q path: softmax over rowmax(H), weighted column-sum of H, broadcast.
// ---------------------------------------------------------------------------
__global__ __launch_bounds__(1024) void c2q_stats_kernel(
    const float* __restrict__ rowb, float* __restrict__ stats,
    float* __restrict__ hrow)
{
    __shared__ float sm[1024], sz[1024];
    const int t = threadIdx.x;
    float vals[8];
    float m = -1e30f;
    #pragma unroll
    for (int i = 0; i < 8; ++i) { vals[i] = rowb[t + i * 1024]; m = fmaxf(m, vals[i]); }
    sm[t] = m; __syncthreads();
    for (int s = 512; s > 0; s >>= 1) { if (t < s) sm[t] = fmaxf(sm[t], sm[t + s]); __syncthreads(); }
    const float M = sm[0];
    float z = 0.f;
    #pragma unroll
    for (int i = 0; i < 8; ++i) z += __expf(vals[i] - M);
    sz[t] = z; __syncthreads();
    for (int s = 512; s > 0; s >>= 1) { if (t < s) sz[t] += sz[t + s]; __syncthreads(); }
    if (t == 0) { stats[0] = M; stats[1] = sz[0]; }
    hrow[t] = 0.f;   // zero the accumulator (t covers 0..1023 == D)
}

__global__ __launch_bounds__(256) void c2q_accum_kernel(
    const float* __restrict__ H, const float* __restrict__ rowb,
    const float* __restrict__ stats, float* __restrict__ hrow)
{
    __shared__ float wsh[256];
    const int d  = blockIdx.x * 256 + threadIdx.x;
    const int r0 = blockIdx.y * 256;
    const float M = stats[0];
    const float rcpZ = 1.0f / stats[1];
    wsh[threadIdx.x] = __expf(rowb[r0 + threadIdx.x] - M) * rcpZ;
    __syncthreads();
    float acc = 0.f;
    for (int i = 0; i < 256; ++i)
        acc += wsh[i] * H[(size_t)(r0 + i) * D_DIM + d];
    atomicAdd(&hrow[d], acc);
}

__global__ __launch_bounds__(256) void broadcast_kernel(
    const float* __restrict__ hrow, float* __restrict__ out2)
{
    const size_t i = ((size_t)blockIdx.x * blockDim.x + threadIdx.x) * 4;
    const int d = (int)(i & (D_DIM - 1));
    *(float4*)(out2 + i) = *(const float4*)(hrow + d);
}

// ---------------------------------------------------------------------------
extern "C" void kernel_launch(void* const* d_in, const int* in_sizes, int n_in,
                              void* d_out, int out_size, void* d_ws, size_t ws_size,
                              hipStream_t stream) {
    (void)in_sizes; (void)n_in; (void)out_size; (void)ws_size;
    const float* H    = (const float*)d_in[0];
    const float* U    = (const float*)d_in[1];
    const float* w_q  = (const float*)d_in[2];
    const float* b_q  = (const float*)d_in[3];
    const float* w_c  = (const float*)d_in[4];
    const float* b_c  = (const float*)d_in[5];
    const float* w_qc = (const float*)d_in[6];
    const float* b_qc = (const float*)d_in[7];
    float* out = (float*)d_out;

    char* ws = (char*)d_ws;
    size_t off = 0;
    auto alloc = [&](size_t bytes) -> void* {
        void* p = ws + off;
        off = (off + bytes + 255) & ~(size_t)255;
        return p;
    };
    unsigned short* Hs   = (unsigned short*)alloc((size_t)C_LEN * D_DIM * 2); // 16 MB
    unsigned short* Ub   = (unsigned short*)alloc((size_t)Q_LEN * D_DIM * 2); //  2 MB
    unsigned short* Ut   = (unsigned short*)alloc((size_t)D_DIM * Q_LEN * 2); //  2 MB
    float*          S    = (float*)         alloc((size_t)C_LEN * Q_LEN * 4); // 32 MB
    unsigned short* P    = (unsigned short*)alloc((size_t)C_LEN * Q_LEN * 2); // 16 MB
    float*          cterm= (float*)alloc(C_LEN * 4);
    float*          rowb = (float*)alloc(C_LEN * 4);
    float*          qterm= (float*)alloc(Q_LEN * 4);
    float*          mq   = (float*)alloc(Q_LEN * 4);
    float*          rzq  = (float*)alloc(Q_LEN * 4);
    float*          hrow = (float*)alloc(D_DIM * 4);
    float*          stats= (float*)alloc(2 * 4);

    // Phase 1: prep / precision conversion
    prep_h_kernel<<<C_LEN / 8, 256, 0, stream>>>(H, w_qc, w_c, b_c, Hs, cterm, rowb);
    prep_u_kernel<<<Q_LEN / 8, 256, 0, stream>>>(U, w_q, b_q, Ub, Ut, qterm);

    // Phase 2: S = Hs @ Ub^T + biases   (WMMA bf16, 64x64 wave tiles)
    gemm_s_kernel<<<dim3(C_LEN / 256, Q_LEN / 128), 256, 0, stream>>>(Hs, Ub, cterm, qterm, b_qc, S);

    // Phase 3: column softmax -> P (bf16)
    colstats_kernel<<<Q_LEN / 16, 256, 0, stream>>>(S, mq, rzq);
    make_p_kernel<<<(C_LEN * Q_LEN) / (256 * 4), 256, 0, stream>>>(S, mq, rzq, P);

    // Phase 4: U_toggler = P @ U   (WMMA bf16) -> first half of d_out
    gemm_out_kernel<<<dim3(C_LEN / 256, D_DIM / 128), 256, 0, stream>>>(P, Ut, out);

    // Phase 5: H_toggler = broadcast(softmax(rowmax(H)) @ H) -> second half
    float* out2 = out + (size_t)C_LEN * D_DIM;
    c2q_stats_kernel<<<1, 1024, 0, stream>>>(rowb, stats, hrow);
    c2q_accum_kernel<<<dim3(D_DIM / 256, C_LEN / 256), 256, 0, stream>>>(H, rowb, stats, hrow);
    broadcast_kernel<<<(C_LEN * D_DIM) / (256 * 4), 256, 0, stream>>>(hrow, out2);
}
